// EdgeDecoder_24481313587756
// MI455X (gfx1250) — matrix operand order repaired and checked by reading, not compiled, not run on hardware
//
#include <hip/hip_runtime.h>
#include <hip/hip_bf16.h>

// ---------------------------------------------------------------------------
// EdgeDecoder for MI455X (gfx1250): bf16 WMMA (f32 accumulate), wave32.
//   p = sigmoid( elu( elu([src||dst] @ Wbi + b) @ W1 + b1 ) @ W2 + b2 )
// Layer-3 dot is fused into the layer-2 epilogue via in-wave shfl_xor
// butterflies (no serial LDS dot, no float atomics -> deterministic).
// ---------------------------------------------------------------------------

typedef __bf16 bf16;
typedef __attribute__((ext_vector_type(16))) __bf16 v16bf;
typedef __attribute__((ext_vector_type(8)))  __bf16 v8bf;
typedef __attribute__((ext_vector_type(4)))  __bf16 v4bf;
typedef __attribute__((ext_vector_type(8)))  float  v8f;

#define E_EDGES 300000
#define DIM     128
#define TWO_D   256
#define HID     128
#define MT      32                 // edges per tile (two 16-row WMMA tiles)
#define NTILES  (E_EDGES / MT)     // 9375 exactly, no tail
#define ES      264                // e_lds row stride (bf16), padded vs bank conflicts
#define HS      136                // h_lds row stride (bf16), padded

// LDS carve (bytes)
#define OFF_WBI   0                // 8 ntile * 8 kstep * 32 lane * 16 bf16 = 64 KiB
#define OFF_W1    65536            // 8 ntile * 4 kstep * 32 lane * 16 bf16 = 32 KiB
#define OFF_E     98304            // 32 * 264 * 2  = 16896
#define OFF_H     115200           // 32 * 136 * 2  = 8704
#define OFF_RED   123904           // 8 waves * 32 rows * 4 = 1024
#define OFF_BBI   124928           // 128 f32
#define OFF_B1    125440           // 128 f32
#define OFF_W2    125952           // 128 f32
#define SMEM_BYTES 126464

__device__ __forceinline__ v8f wmma_bf16(v16bf a, v16bf b, v8f c) {
  // (neg_a, A, neg_b, B, c_mod, C, reuse_a, reuse_b)
  return __builtin_amdgcn_wmma_f32_16x16x32_bf16(false, a, false, b, (short)0, c,
                                                 false, false);
}

__device__ __forceinline__ float eluf(float x) {
  return x > 0.0f ? x : (__expf(x) - 1.0f);
}

__global__ void __launch_bounds__(256)
edge_decoder_kernel(const float* __restrict__ user_emb,
                    const float* __restrict__ item_emb,
                    const int*   __restrict__ ei_ui,
                    const int*   __restrict__ ei_iu,
                    const float* __restrict__ Wbi_ui, const float* __restrict__ bbi_ui,
                    const float* __restrict__ W1_ui,  const float* __restrict__ b1_ui,
                    const float* __restrict__ W2_ui,  const float* __restrict__ b2_ui,
                    const float* __restrict__ Wbi_iu, const float* __restrict__ bbi_iu,
                    const float* __restrict__ W1_iu,  const float* __restrict__ b1_iu,
                    const float* __restrict__ W2_iu,  const float* __restrict__ b2_iu,
                    float* __restrict__ out) {
  extern __shared__ __align__(128) char smem[];
  bf16*  Wbi_sw = (bf16*)(smem + OFF_WBI);
  bf16*  W1_sw  = (bf16*)(smem + OFF_W1);
  bf16*  e_lds  = (bf16*)(smem + OFF_E);
  bf16*  h_lds  = (bf16*)(smem + OFF_H);
  float* redL   = (float*)(smem + OFF_RED);
  float* bbiL   = (float*)(smem + OFF_BBI);
  float* b1L    = (float*)(smem + OFF_B1);
  float* w2L    = (float*)(smem + OFF_W2);

  const int ty = blockIdx.y;                       // 0: user->item, 1: item->user
  const float* srcE = ty ? item_emb : user_emb;
  const float* dstE = ty ? user_emb : item_emb;
  const int*   ei   = ty ? ei_iu    : ei_ui;
  const float* Wbi  = ty ? Wbi_iu   : Wbi_ui;
  const float* bbi  = ty ? bbi_iu   : bbi_ui;
  const float* W1   = ty ? W1_iu    : W1_ui;
  const float* b1   = ty ? b1_iu    : b1_ui;
  const float* W2   = ty ? W2_iu    : W2_ui;
  const float* b2   = ty ? b2_iu    : b2_ui;

  // ---- Preamble: stage weights in LDS, pre-swizzled into B-fragment layout.
  // B fragment (16x16x32 bf16): lane = half*16 + n; lane holds k_local =
  // half*8 + j (j<8), 16 + half*8 + (j-8) (j>=8) -> 16 bf16 per (tile,lane).
  for (int idx = threadIdx.x; idx < 8 * 8 * 32 * 16; idx += 256) {
    int j  = idx & 15;
    int ln = (idx >> 4) & 31;
    int ks = (idx >> 9) & 7;
    int nt = idx >> 12;
    int k  = ks * 32 + (j & 8) * 2 + (ln >> 4) * 8 + (j & 7);
    int n  = nt * 16 + (ln & 15);
    Wbi_sw[idx] = (bf16)Wbi[k * HID + n];
  }
  for (int idx = threadIdx.x; idx < 8 * 4 * 32 * 16; idx += 256) {
    int j  = idx & 15;
    int ln = (idx >> 4) & 31;
    int ks = (idx >> 9) & 3;
    int nt = idx >> 11;
    int k  = ks * 32 + (j & 8) * 2 + (ln >> 4) * 8 + (j & 7);
    int n  = nt * 16 + (ln & 15);
    W1_sw[idx] = (bf16)W1[k * HID + n];
  }
  for (int i = threadIdx.x; i < HID; i += 256) {
    bbiL[i] = bbi[i];
    b1L[i]  = b1[i];
    w2L[i]  = W2[i];
  }
  __syncthreads();

  const int lane = threadIdx.x & 31;
  const int wv   = threadIdx.x >> 5;      // 8 waves; wave owns N-tile wv
  const int hh   = lane >> 4;             // lane half (K split / row-half of C)
  const int l15  = lane & 15;
  const int nCol = wv * 16 + l15;         // global hidden column for C frag

  for (int tile = blockIdx.x; tile < NTILES; tile += gridDim.x) {
    // ---- Gather 32 edges: [src(128) || dst(128)] fp32 -> bf16 in LDS.
    {
      int m = threadIdx.x >> 3;           // 0..31 (edge row)
      int g = threadIdx.x & 7;            // 8 threads per row
      size_t e = (size_t)tile * MT + m;
      const float* srow = srcE + (size_t)ei[e] * DIM;
      const float* drow = dstE + (size_t)ei[(size_t)E_EDGES + e] * DIM;
      bf16* erow = e_lds + m * ES;
#pragma unroll
      for (int c = 0; c < 4; ++c) {
        int col = g * 4 + c * 32;
        float4 v = *(const float4*)(srow + col);
        v4bf pv = {(bf16)v.x, (bf16)v.y, (bf16)v.z, (bf16)v.w};
        *(v4bf*)(erow + col) = pv;
        float4 u = *(const float4*)(drow + col);
        v4bf pu = {(bf16)u.x, (bf16)u.y, (bf16)u.z, (bf16)u.w};
        *(v4bf*)(erow + DIM + col) = pu;
      }
    }
    __syncthreads();

    // ---- Layer 1: h = elu(e[32x256] @ Wbi[256x128] + b). K = 8 x 32.
    {
      v8f a0, a1;
      float bias = bbiL[nCol];
#pragma unroll
      for (int r = 0; r < 8; ++r) { a0[r] = bias; a1[r] = bias; }
#pragma unroll
      for (int ks = 0; ks < 8; ++ks) {
        v16bf bfrag = *(const v16bf*)(Wbi_sw + (((wv * 8 + ks) * 32 + lane) << 4));
        union { v16bf v; v8bf h[2]; } fa0, fa1;
        const bf16* p0 = e_lds + l15 * ES + ks * 32 + hh * 8;
        fa0.h[0] = *(const v8bf*)(p0);
        fa0.h[1] = *(const v8bf*)(p0 + 16);
        const bf16* p1 = p0 + 16 * ES;
        fa1.h[0] = *(const v8bf*)(p1);
        fa1.h[1] = *(const v8bf*)(p1 + 16);
        a0 = wmma_bf16(fa0.v, bfrag, a0);
        a1 = wmma_bf16(fa1.v, bfrag, a1);
      }
#pragma unroll
      for (int r = 0; r < 8; ++r) {
        int m0 = r + 8 * hh;              // C frag: M = r + 8*half, N = nCol
        h_lds[m0 * HS + nCol]        = (bf16)eluf(a0[r]);
        h_lds[(m0 + 16) * HS + nCol] = (bf16)eluf(a1[r]);
      }
    }
    __syncthreads();

    // ---- Layer 2 + fused layer-3 partials.
    {
      v8f a0, a1;
      float bias = b1L[nCol];
#pragma unroll
      for (int r = 0; r < 8; ++r) { a0[r] = bias; a1[r] = bias; }
#pragma unroll
      for (int ks = 0; ks < 4; ++ks) {
        v16bf bfrag = *(const v16bf*)(W1_sw + (((wv * 4 + ks) * 32 + lane) << 4));
        union { v16bf v; v8bf h[2]; } fa0, fa1;
        const bf16* p0 = h_lds + l15 * HS + ks * 32 + hh * 8;
        fa0.h[0] = *(const v8bf*)(p0);
        fa0.h[1] = *(const v8bf*)(p0 + 16);
        const bf16* p1 = p0 + 16 * HS;
        fa1.h[0] = *(const v8bf*)(p1);
        fa1.h[1] = *(const v8bf*)(p1 + 16);
        a0 = wmma_bf16(fa0.v, bfrag, a0);
        a1 = wmma_bf16(fa1.v, bfrag, a1);
      }

      // h2[m][nCol] = elu(acc); contribute h2 * w2[nCol] to row sums.
      // Butterfly over the 16-lane halves: masks 1,2,4,8 stay inside each
      // half, and the two halves carry disjoint row sets -> one reduction
      // tree handles both.
      float w2v = w2L[nCol];
      float s0[8], s1[8];
#pragma unroll
      for (int r = 0; r < 8; ++r) {
        s0[r] = eluf(a0[r]) * w2v;
        s1[r] = eluf(a1[r]) * w2v;
      }
#pragma unroll
      for (int mask = 1; mask < 16; mask <<= 1) {
#pragma unroll
        for (int r = 0; r < 8; ++r) {
          s0[r] += __shfl_xor(s0[r], mask, 32);
          s1[r] += __shfl_xor(s1[r], mask, 32);
        }
      }
      if (l15 == 0) {                      // lanes 0 and 16 hold the half-sums
        float* rw = redL + wv * 32 + hh * 8;
        float4 p0 = {s0[0], s0[1], s0[2], s0[3]};
        float4 p1 = {s0[4], s0[5], s0[6], s0[7]};
        float4 p2 = {s1[0], s1[1], s1[2], s1[3]};
        float4 p3 = {s1[4], s1[5], s1[6], s1[7]};
        *(float4*)(rw)      = p0;          // rows  8*hh .. 8*hh+7
        *(float4*)(rw + 4)  = p1;
        *(float4*)(rw + 16) = p2;          // rows 16+8*hh .. 16+8*hh+7
        *(float4*)(rw + 20) = p3;
      }
    }
    __syncthreads();

    // ---- Final: sum 8 wave-partials per edge, sigmoid, store.
    if (threadIdx.x < MT) {
      int m = threadIdx.x;
      float acc = b2[0];
#pragma unroll
      for (int s = 0; s < 8; ++s) acc += redL[s * 32 + m];
      out[(size_t)ty * E_EDGES + (size_t)tile * MT + m] =
          1.0f / (1.0f + __expf(-acc));
    }
    // e_lds rewrite next iter is fenced by the post-gather barrier; redL is
    // next written only after two further barriers.
  }
}

extern "C" void kernel_launch(void* const* d_in, const int* in_sizes, int n_in,
                              void* d_out, int out_size, void* d_ws, size_t ws_size,
                              hipStream_t stream) {
  (void)in_sizes; (void)n_in; (void)out_size; (void)d_ws; (void)ws_size;

  const float* user_emb = (const float*)d_in[0];
  const float* item_emb = (const float*)d_in[1];
  const int*   ei_ui    = (const int*)d_in[2];
  const int*   ei_iu    = (const int*)d_in[3];
  const float* Wbi_ui   = (const float*)d_in[4];
  const float* bbi_ui   = (const float*)d_in[5];
  const float* W1_ui    = (const float*)d_in[6];
  const float* b1_ui    = (const float*)d_in[7];
  const float* W2_ui    = (const float*)d_in[8];
  const float* b2_ui    = (const float*)d_in[9];
  const float* Wbi_iu   = (const float*)d_in[10];
  const float* bbi_iu   = (const float*)d_in[11];
  const float* W1_iu    = (const float*)d_in[12];
  const float* b1_iu    = (const float*)d_in[13];
  const float* W2_iu    = (const float*)d_in[14];
  const float* b2_iu    = (const float*)d_in[15];
  float* out = (float*)d_out;

  hipFuncSetAttribute(reinterpret_cast<const void*>(edge_decoder_kernel),
                      hipFuncAttributeMaxDynamicSharedMemorySize, SMEM_BYTES);

  dim3 grid(512, 2, 1);   // y = edge type; x grid-strides over 9375 tiles
  dim3 block(256, 1, 1);  // 8 wave32 waves
  edge_decoder_kernel<<<grid, block, SMEM_BYTES, stream>>>(
      user_emb, item_emb, ei_ui, ei_iu,
      Wbi_ui, bbi_ui, W1_ui, b1_ui, W2_ui, b2_ui,
      Wbi_iu, bbi_iu, W1_iu, b1_iu, W2_iu, b2_iu, out);
}
